// ProtoDINO_90134183674041
// MI455X (gfx1250) — matrix operand order repaired and checked by reading, not compiled, not run on hardware
//
#include <hip/hip_runtime.h>
#include <hip/hip_bf16.h>
#include <math.h>

typedef __bf16 bf16;
typedef __attribute__((ext_vector_type(16))) bf16 v16bf;
typedef __attribute__((ext_vector_type(8)))  bf16 bf16x8;
typedef __attribute__((ext_vector_type(8)))  float v8f;
typedef __attribute__((ext_vector_type(4)))  unsigned u32x4;
typedef __attribute__((ext_vector_type(8)))  int      i32x8;
typedef __attribute__((ext_vector_type(4)))  int      i32x4;

#define M_TOK   43808      // B*NP
#define DIM_    768
#define NCOL    1005       // C*K
#define NPADB   1008
#define NCLS_   200
#define CC      201
#define KK      5
#define BB      32
#define NP_     1369
#define EPSF    1e-8f
#define GAMMA_  0.99f
#define NKT     (DIM_ / 32)

#if __has_builtin(__builtin_amdgcn_tensor_load_to_lds)
#define USE_TDM 1
#else
#define USE_TDM 0
#endif

// ---------------- helpers ----------------
__device__ __forceinline__ float waveRed(float v) {
  #pragma unroll
  for (int o = 16; o > 0; o >>= 1) v += __shfl_xor(v, o, 32);
  return v;
}

__global__ void k_zerof(float* p, int n) {
  int i = blockIdx.x * 256 + threadIdx.x;
  if (i < n) p[i] = 0.0f;
}

// ---------------- row norms ----------------
__global__ void k_tok_norm(const float* __restrict__ X, float* __restrict__ invx) {
  int row = blockIdx.x * 8 + (threadIdx.x >> 5);
  int lane = threadIdx.x & 31;
  if (row >= M_TOK) return;
  const float* r = X + (size_t)row * DIM_;
  float ss = 0.f;
  for (int d = lane; d < DIM_; d += 32) { float x = r[d]; ss += x * x; }
  ss = waveRed(ss);
  if (!lane) invx[row] = 1.0f / fmaxf(sqrtf(ss), EPSF);
}

__global__ void k_pro_norm(const float* __restrict__ P, float* __restrict__ invp) {
  int row = blockIdx.x * 8 + (threadIdx.x >> 5);
  int lane = threadIdx.x & 31;
  if (row >= NPADB) return;
  if (row >= NCOL) { if (!lane) invp[row] = 0.0f; return; }
  const float* r = P + (size_t)row * DIM_;
  float ss = 0.f;
  for (int d = lane; d < DIM_; d += 32) { float x = r[d]; ss += x * x; }
  ss = waveRed(ss);
  if (!lane) invp[row] = 1.0f / fmaxf(sqrtf(ss), EPSF);
}

// ---------------- normalized bf16 staging (halves GEMM-side traffic) ----------------
__global__ void k_cvt_tok(const float* __restrict__ X, const float* __restrict__ invx,
                          bf16* __restrict__ out) {
  size_t i = (size_t)blockIdx.x * 256 + threadIdx.x;
  if (i >= (size_t)M_TOK * DIM_) return;
  int row = (int)(i / DIM_);
  __builtin_prefetch(&X[i + 8192], 0, 1);
  out[i] = (bf16)(X[i] * invx[row]);
}
__global__ void k_cvt_pro(const float* __restrict__ P, const float* __restrict__ invp,
                          bf16* __restrict__ out) {
  size_t i = (size_t)blockIdx.x * 256 + threadIdx.x;
  if (i >= (size_t)NPADB * DIM_) return;
  int row = (int)(i / DIM_);
  out[i] = (row < NCOL) ? (bf16)(P[i] * invp[row]) : (bf16)0.0f;
}

// ---------------- TDM descriptor helper (ISA 8.3/8.4, 2-D tensor) ----------------
#if USE_TDM
__device__ __forceinline__ void tdm_load_2d(unsigned lds_off, const bf16* gptr,
                                            unsigned tensor_d0, unsigned tensor_d1,
                                            unsigned tile_d0, unsigned tile_d1,
                                            unsigned stride0) {
  unsigned long long ga = (unsigned long long)(uintptr_t)gptr;
  u32x4 g0;
  g0[0] = 1u;                                            // count=1, user descriptor
  g0[1] = lds_off;                                       // lds_addr (bytes)
  g0[2] = (unsigned)(ga & 0xFFFFFFFFull);                // global_addr[31:0]
  g0[3] = (unsigned)((ga >> 32) & 0x1FFFFFFull) | (2u << 30);  // addr[56:32] | type=2
  i32x8 g1;
  g1[0] = (int)(1u << 16);                               // wg_mask=0, data_size=1 (2B)
  g1[1] = (int)((tensor_d0 & 0xFFFFu) << 16);            // tensor_dim0[15:0] @bits63:48
  g1[2] = (int)(((tensor_d0 >> 16) & 0xFFFFu) | ((tensor_d1 & 0xFFFFu) << 16));
  g1[3] = (int)(((tensor_d1 >> 16) & 0xFFFFu) | ((tile_d0 & 0xFFFFu) << 16));
  g1[4] = (int)(tile_d1 & 0xFFFFu);                      // tile_dim1; tile_dim2=0
  g1[5] = (int)stride0;                                  // tensor_dim0_stride[31:0]
  g1[6] = 0;                                             // stride msbs / dim1_stride
  g1[7] = 0;
  i32x4 z4 = {0, 0, 0, 0};
  i32x8 z8 = {0, 0, 0, 0, 0, 0, 0, 0};
  __builtin_amdgcn_tensor_load_to_lds(g0, g1, z4, z4, z8, 0);   // 6-arg toolchain form
}
#endif

// ---------------- main WMMA GEMM: logits = scale * tokn @ pron^T ----------------
#if USE_TDM
__global__ __launch_bounds__(256) void k_gemm(
    const bf16* __restrict__ Abf, const bf16* __restrict__ Bbf,
    const float* __restrict__ scale, float* __restrict__ out)
{
  __shared__ __align__(16) bf16 As[2][128 * 32];
  __shared__ __align__(16) bf16 Bs[2][64 * 32];
  const int tid  = threadIdx.x;
  const int lane = tid & 31, wave = tid >> 5;
  const int wm = wave & 3, wn = wave >> 2;          // 4x2 wave grid
  const int rowBase = blockIdx.y * 128;
  const int colBase = blockIdx.x * 64;
  const int half = lane >> 4, l16 = lane & 15;
  const float sc = scale[0];
  v8f acc[2][2] = {};

  // pre-issue tile 0 (wave 0 only; TDM ignores EXEC, OOB rows read as zero)
  if (tid < 32) {
    tdm_load_2d((unsigned)(uintptr_t)&As[0][0], Abf + (size_t)rowBase * DIM_,
                DIM_, M_TOK, 32, 128, DIM_);
    tdm_load_2d((unsigned)(uintptr_t)&Bs[0][0], Bbf + (size_t)colBase * DIM_,
                DIM_, NPADB, 32, 64, DIM_);
  }

  for (int kt = 0; kt < NKT; ++kt) {
    const int buf = kt & 1;
    if (tid < 32) {
      if (kt + 1 < NKT) {   // double-buffer: overlap next DMA with this tile's WMMA
        tdm_load_2d((unsigned)(uintptr_t)&As[buf ^ 1][0],
                    Abf + (size_t)rowBase * DIM_ + (kt + 1) * 32,
                    DIM_, M_TOK, 32, 128, DIM_);
        tdm_load_2d((unsigned)(uintptr_t)&Bs[buf ^ 1][0],
                    Bbf + (size_t)colBase * DIM_ + (kt + 1) * 32,
                    DIM_, NPADB, 32, 64, DIM_);
        __builtin_amdgcn_s_wait_tensorcnt(2);   // current tile complete
      } else {
        __builtin_amdgcn_s_wait_tensorcnt(0);
      }
    }
    __syncthreads();

    // ---- fragments per documented 16-bit layouts ----
    v16bf afr[2], bfr[2];
    #pragma unroll
    for (int tm = 0; tm < 2; ++tm) {
      int rl = wm * 32 + tm * 16 + l16;
      bf16x8 lo = *(const bf16x8*)&As[buf][rl * 32 + half * 8];
      bf16x8 hi = *(const bf16x8*)&As[buf][rl * 32 + 16 + half * 8];
      #pragma unroll
      for (int j = 0; j < 8; ++j) { afr[tm][j] = lo[j]; afr[tm][j + 8] = hi[j]; }
    }
    #pragma unroll
    for (int tn = 0; tn < 2; ++tn) {
      int nl = wn * 32 + tn * 16 + l16;
      bf16x8 lo = *(const bf16x8*)&Bs[buf][nl * 32 + half * 8];
      bf16x8 hi = *(const bf16x8*)&Bs[buf][nl * 32 + 16 + half * 8];
      #pragma unroll
      for (int j = 0; j < 8; ++j) { bfr[tn][j] = lo[j]; bfr[tn][j + 8] = hi[j]; }
    }
    #pragma unroll
    for (int tm = 0; tm < 2; ++tm)
      #pragma unroll
      for (int tn = 0; tn < 2; ++tn)
        acc[tm][tn] = __builtin_amdgcn_wmma_f32_16x16x32_bf16(
            false, afr[tm], false, bfr[tn], (short)0, acc[tm][tn], false, false);
    __syncthreads();
  }

  #pragma unroll
  for (int tm = 0; tm < 2; ++tm)
    #pragma unroll
    for (int tn = 0; tn < 2; ++tn)
      #pragma unroll
      for (int i = 0; i < 8; ++i) {
        int row = rowBase + wm * 32 + tm * 16 + half * 8 + i;
        int col = colBase + wn * 32 + tn * 16 + l16;
        if (row < M_TOK && col < NCOL)
          out[(size_t)row * NCOL + col] = sc * acc[tm][tn][i];
      }
}
#else   // fallback: manual LDS staging
__global__ __launch_bounds__(256) void k_gemm(
    const bf16* __restrict__ Abf, const bf16* __restrict__ Bbf,
    const float* __restrict__ scale, float* __restrict__ out)
{
  __shared__ __align__(16) bf16 As[128 * 32];
  __shared__ __align__(16) bf16 Bs[64 * 32];
  const int tid  = threadIdx.x;
  const int lane = tid & 31, wave = tid >> 5;
  const int wm = wave & 3, wn = wave >> 2;
  const int rowBase = blockIdx.y * 128;
  const int colBase = blockIdx.x * 64;
  const int half = lane >> 4, l16 = lane & 15;
  const float sc = scale[0];
  v8f acc[2][2] = {};

  for (int kt = 0; kt < NKT; ++kt) {
    const int kb = kt * 32;
    #pragma unroll
    for (int it = 0; it < 2; ++it) {
      int c  = tid + it * 256;
      int rl = c >> 2, cl = (c & 3) * 8;
      int gr = rowBase + rl;
      bf16x8 w;
      if (gr < M_TOK) {
        w = *(const bf16x8*)(Abf + (size_t)gr * DIM_ + kb + cl);
      } else {
        #pragma unroll
        for (int j = 0; j < 8; ++j) w[j] = (bf16)0.0f;
      }
      *(bf16x8*)&As[rl * 32 + cl] = w;
    }
    {
      int rl = tid >> 2, cl = (tid & 3) * 8;
      bf16x8 w = *(const bf16x8*)(Bbf + (size_t)(colBase + rl) * DIM_ + kb + cl);
      *(bf16x8*)&Bs[rl * 32 + cl] = w;
    }
    __syncthreads();
    v16bf afr[2], bfr[2];
    #pragma unroll
    for (int tm = 0; tm < 2; ++tm) {
      int rl = wm * 32 + tm * 16 + l16;
      bf16x8 lo = *(const bf16x8*)&As[rl * 32 + half * 8];
      bf16x8 hi = *(const bf16x8*)&As[rl * 32 + 16 + half * 8];
      #pragma unroll
      for (int j = 0; j < 8; ++j) { afr[tm][j] = lo[j]; afr[tm][j + 8] = hi[j]; }
    }
    #pragma unroll
    for (int tn = 0; tn < 2; ++tn) {
      int nl = wn * 32 + tn * 16 + l16;
      bf16x8 lo = *(const bf16x8*)&Bs[nl * 32 + half * 8];
      bf16x8 hi = *(const bf16x8*)&Bs[nl * 32 + 16 + half * 8];
      #pragma unroll
      for (int j = 0; j < 8; ++j) { bfr[tn][j] = lo[j]; bfr[tn][j + 8] = hi[j]; }
    }
    #pragma unroll
    for (int tm = 0; tm < 2; ++tm)
      #pragma unroll
      for (int tn = 0; tn < 2; ++tn)
        acc[tm][tn] = __builtin_amdgcn_wmma_f32_16x16x32_bf16(
            false, afr[tm], false, bfr[tn], (short)0, acc[tm][tn], false, false);
    __syncthreads();
  }
  #pragma unroll
  for (int tm = 0; tm < 2; ++tm)
    #pragma unroll
    for (int tn = 0; tn < 2; ++tn)
      #pragma unroll
      for (int i = 0; i < 8; ++i) {
        int row = rowBase + wm * 32 + tm * 16 + half * 8 + i;
        int col = colBase + wn * 32 + tn * 16 + l16;
        if (row < M_TOK && col < NCOL)
          out[(size_t)row * NCOL + col] = sc * acc[tm][tn][i];
      }
}
#endif

// ---------------- PCA (power iteration) ----------------
__global__ void k_colsum(const float* __restrict__ X, float* __restrict__ mean) {
  int d = blockIdx.x * 256 + threadIdx.x;
  if (d >= DIM_) return;
  int n0 = blockIdx.y * (M_TOK / 16), n1 = n0 + M_TOK / 16;
  float s = 0.f;
  for (int n = n0; n < n1; ++n) s += X[(size_t)n * DIM_ + d];
  atomicAdd(&mean[d], s);
}

__global__ void k_mean_div_initv(float* mean, float* v) {
  for (int d = threadIdx.x; d < DIM_; d += 256) {
    mean[d] *= (1.0f / (float)M_TOK);
    float t = sinf((float)(d + 1) * 12.9898f) * 43758.5453f;
    v[d] = t - floorf(t) - 0.5f;
  }
}

__global__ void k_pca_prep(const float* mean, const float* v, float* vtmp, float* scal) {
  __shared__ float red[256];
  float s = 0.f;
  for (int d = threadIdx.x; d < DIM_; d += 256) { s += mean[d] * v[d]; vtmp[d] = 0.f; }
  red[threadIdx.x] = s; __syncthreads();
  for (int o = 128; o > 0; o >>= 1) {
    if (threadIdx.x < o) red[threadIdx.x] += red[threadIdx.x + o];
    __syncthreads();
  }
  if (!threadIdx.x) { scal[0] = red[0]; scal[1] = 0.f; }
}

__global__ void k_matvec_u(const float* __restrict__ X, const float* __restrict__ v,
                           float* __restrict__ scal, float* __restrict__ u) {
  int row = blockIdx.x * 8 + (threadIdx.x >> 5);
  int lane = threadIdx.x & 31;
  if (row >= M_TOK) return;
  const float* r = X + (size_t)row * DIM_;
  float s = 0.f;
  for (int d = lane; d < DIM_; d += 32) s += r[d] * v[d];
  s = waveRed(s);
  if (!lane) { float uu = s - scal[0]; u[row] = uu; atomicAdd(&scal[1], uu); }
}

__global__ void k_matvec_vt(const float* __restrict__ X, const float* __restrict__ u,
                            float* __restrict__ vtmp) {
  int d = blockIdx.x * 256 + threadIdx.x;
  if (d >= DIM_) return;
  int n0 = blockIdx.y * (M_TOK / 16), n1 = n0 + M_TOK / 16;
  float s = 0.f;
  for (int n = n0; n < n1; ++n) s += X[(size_t)n * DIM_ + d] * u[n];
  atomicAdd(&vtmp[d], s);
}

__global__ void k_v_finish(const float* mean, const float* vtmp, const float* scal, float* v) {
  __shared__ float red[256];
  __shared__ float nrm;
  float su = scal[1];
  float t[3]; float s = 0.f;
  #pragma unroll
  for (int i = 0; i < 3; ++i) {
    int d = threadIdx.x + i * 256;
    t[i] = vtmp[d] - mean[d] * su;
    s += t[i] * t[i];
  }
  red[threadIdx.x] = s; __syncthreads();
  for (int o = 128; o > 0; o >>= 1) {
    if (threadIdx.x < o) red[threadIdx.x] += red[threadIdx.x + o];
    __syncthreads();
  }
  if (!threadIdx.x) nrm = fmaxf(sqrtf(red[0]), EPSF);
  __syncthreads();
  #pragma unroll
  for (int i = 0; i < 3; ++i) v[threadIdx.x + i * 256] = t[i] / nrm;
}

// ---------------- pseudo labels ----------------
__global__ void k_minmax_init(unsigned* uscal) { uscal[0] = 0xFFFFFFFFu; uscal[1] = 0u; }
__global__ void k_minmax(const float* u, unsigned* uscal) {
  int n = blockIdx.x * 256 + threadIdx.x;
  if (n >= M_TOK) return;
  unsigned b = __float_as_uint(u[n]);
  unsigned e = (b & 0x80000000u) ? ~b : (b | 0x80000000u);
  atomicMin(&uscal[0], e); atomicMax(&uscal[1], e);
}
__global__ void k_thr(const unsigned* uscal, float* scal) {
  unsigned e0 = uscal[0], e1 = uscal[1];
  float mn = __uint_as_float((e0 & 0x80000000u) ? (e0 & 0x7FFFFFFFu) : ~e0);
  float mx = __uint_as_float((e1 & 0x80000000u) ? (e1 & 0x7FFFFFFFu) : ~e1);
  scal[3] = 0.5f * (mn + mx);
}
__global__ void k_pseudo(const float* u, const int* labels, const float* scal,
                         int* labint, float* outPse) {
  int n = blockIdx.x * 256 + threadIdx.x;
  if (n >= M_TOK) return;
  int b = n / NP_;
  int lab = (u[n] <= scal[3]) ? labels[b] : (CC - 1);
  labint[n] = lab;
  outPse[n] = (float)lab;
}

// ---------------- sinkhorn (mask-collapsed: q is [N,K] at c = lab[n]) ----------------
__global__ void k_cls_zero(float* Nc, float* Stot, float* colsum, int* flags) {
  for (int i = threadIdx.x; i < CC; i += 256) { Nc[i] = 0.f; Stot[i] = 0.f; flags[i] = 0; }
  for (int i = threadIdx.x; i < CC * KK; i += 256) colsum[i] = 0.f;
}
__global__ void k_flags(const int* labels, int* flags) {
  if (threadIdx.x < BB) flags[labels[threadIdx.x]] = 1;
}
__global__ void k_qinit(const float* __restrict__ logits, const int* __restrict__ labint,
                        float* __restrict__ q, float* Stot, float* Nc) {
  int n = blockIdx.x * 256 + threadIdx.x;
  if (n >= M_TOK) return;
  int lab = labint[n];
  const float* lr = logits + (size_t)n * NCOL + lab * KK;
  float s = 0.f;
  #pragma unroll
  for (int k = 0; k < KK; ++k) { float e = expf(lr[k]); q[n * KK + k] = e; s += e; }
  atomicAdd(&Stot[lab], s);
  atomicAdd(&Nc[lab], 1.0f);
}
__global__ void k_qscale(float* q, const int* labint, const float* Stot) {
  int n = blockIdx.x * 256 + threadIdx.x;
  if (n >= M_TOK) return;
  float inv = 1.0f / fmaxf(Stot[labint[n]], EPSF);
  #pragma unroll
  for (int k = 0; k < KK; ++k) q[n * KK + k] *= inv;
}
__global__ void k_zerocol(float* colsum) {
  for (int i = threadIdx.x; i < CC * KK; i += 256) colsum[i] = 0.f;
}
__global__ void k_colsumq(const float* q, const int* labint, float* colsum) {
  int n = blockIdx.x * 256 + threadIdx.x;
  if (n >= M_TOK) return;
  int lab = labint[n];
  #pragma unroll
  for (int k = 0; k < KK; ++k) atomicAdd(&colsum[lab * KK + k], q[n * KK + k]);
}
__global__ void k_qrow(float* q, const int* labint, const float* colsum, const float* Nc) {
  int n = blockIdx.x * 256 + threadIdx.x;
  if (n >= M_TOK) return;
  int lab = labint[n];
  float t[KK], r = 0.f;
  #pragma unroll
  for (int k = 0; k < KK; ++k) {
    t[k] = q[n * KK + k] / (fmaxf(colsum[lab * KK + k], EPSF) * (float)KK);
    r += t[k];
  }
  float inv = 1.0f / (fmaxf(r, EPSF) * fmaxf(Nc[lab], 1.0f));
  #pragma unroll
  for (int k = 0; k < KK; ++k) q[n * KK + k] = t[k] * inv;
}
__global__ void k_qmul(float* q, const int* labint, const float* Nc) {
  int n = blockIdx.x * 256 + threadIdx.x;
  if (n >= M_TOK) return;
  float nc = Nc[labint[n]];
  #pragma unroll
  for (int k = 0; k < KK; ++k) q[n * KK + k] *= nc;
}

// ---------------- prototype update ----------------
__global__ void k_pnew(const float* __restrict__ X, const float* __restrict__ invx,
                       const int* __restrict__ labint, const float* __restrict__ q,
                       float* __restrict__ Pn) {
  int row = blockIdx.x * 8 + (threadIdx.x >> 5);
  int lane = threadIdx.x & 31;
  if (row >= M_TOK) return;
  int lab = labint[row];
  float inv = invx[row];
  float qv[KK];
  #pragma unroll
  for (int k = 0; k < KK; ++k) qv[k] = q[row * KK + k];
  const float* r = X + (size_t)row * DIM_;
  float* base = Pn + (size_t)lab * KK * DIM_;
  for (int d = lane; d < DIM_; d += 32) {
    float t = r[d] * inv;
    #pragma unroll
    for (int k = 0; k < KK; ++k) atomicAdd(&base[k * DIM_ + d], qv[k] * t);
  }
}
__global__ void k_proto_out(const float* proto, const float* Pn, const int* flags, float* out) {
  int i = blockIdx.x * 256 + threadIdx.x;
  if (i >= CC * KK * DIM_) return;
  int c = i / (KK * DIM_);
  float pv = proto[i];
  out[i] = flags[c] ? (GAMMA_ * pv + (1.0f - GAMMA_) * Pn[i]) : pv;
}

// ---------------- pooled heads ----------------
__global__ void k_imglog(const float* __restrict__ logits, float* __restrict__ img) {
  int j = blockIdx.x * 256 + threadIdx.x;
  if (j >= NCOL) return;
  int b = blockIdx.y;
  const float* base = logits + (size_t)b * NP_ * NCOL + j;
  float s = 0.f;
  for (int p = 0; p < NP_; ++p) s += base[(size_t)p * NCOL];
  img[b * NCOL + j] = s * (1.0f / (float)NP_);
}
__global__ void k_clslog(const float* __restrict__ img, const float* __restrict__ sa,
                         float* __restrict__ out) {
  int c = threadIdx.x;
  if (c >= NCLS_) return;
  int b = blockIdx.x;
  float sv[KK], m = -1e30f;
  #pragma unroll
  for (int k = 0; k < KK; ++k) { sv[k] = sa[c * KK + k]; m = fmaxf(m, sv[k]); }
  float e[KK], es = 0.f;
  #pragma unroll
  for (int k = 0; k < KK; ++k) { e[k] = expf(sv[k] - m); es += e[k]; }
  const float* row = img + b * NCOL + c * KK;
  float s = 0.f;
  #pragma unroll
  for (int k = 0; k < KK; ++k) s += row[k] * (e[k] / es * (float)KK);
  out[b * NCLS_ + c] = s;
}

// ---------------- launch ----------------
extern "C" void kernel_launch(void* const* d_in, const int* in_sizes, int n_in,
                              void* d_out, int out_size, void* d_ws, size_t ws_size,
                              hipStream_t stream) {
  const float* X      = (const float*)d_in[0];   // patch_tokens [32,1369,768]
  const int*   labels = (const int*)d_in[1];     // [32]
  const float* proto  = (const float*)d_in[2];   // [201,5,768]
  const float* sa     = (const float*)d_in[3];   // [200,5]
  const float* scale  = (const float*)d_in[4];   // scalar

  float* out = (float*)d_out;
  float* outLogits = out;                           // 44,027,040
  float* outImg    = out + 44027040ULL;             // 32*1005
  float* outCls    = out + 44059200ULL;             // 32*200
  float* outPse    = out + 44065600ULL;             // 43808
  float* outPro    = out + 44109408ULL;             // 771840

  char* p = (char*)d_ws;
  auto alloc = [&](size_t bytes) -> void* {
    void* r = (void*)p; p += (bytes + 255) & ~(size_t)255; return r;
  };
  float*    invx   = (float*)alloc((size_t)M_TOK * 4);
  float*    invp   = (float*)alloc((size_t)NPADB * 4);
  float*    mean   = (float*)alloc(DIM_ * 4);
  float*    v      = (float*)alloc(DIM_ * 4);
  float*    vtmp   = (float*)alloc(DIM_ * 4);
  float*    u      = (float*)alloc((size_t)M_TOK * 4);
  float*    scal   = (float*)alloc(16 * 4);
  unsigned* uscal  = (unsigned*)alloc(2 * 4);
  float*    q      = (float*)alloc((size_t)M_TOK * KK * 4);
  float*    Nc     = (float*)alloc(CC * 4);
  float*    Stot   = (float*)alloc(CC * 4);
  float*    colsum = (float*)alloc(CC * KK * 4);
  int*      flags  = (int*)alloc(CC * 4);
  int*      labint = (int*)alloc((size_t)M_TOK * 4);
  float*    Pn     = (float*)alloc((size_t)CC * KK * DIM_ * 4);
  bf16*     tokbf  = (bf16*)alloc((size_t)M_TOK * DIM_ * 2);
  bf16*     probf  = (bf16*)alloc((size_t)NPADB * DIM_ * 2);
  (void)ws_size; (void)n_in; (void)in_sizes; (void)out_size;

  // 1) row inverse norms + normalized bf16 staging
  k_tok_norm<<<M_TOK / 8, 256, 0, stream>>>(X, invx);
  k_pro_norm<<<NPADB / 8, 256, 0, stream>>>(proto, invp);
  k_cvt_tok<<<(int)(((size_t)M_TOK * DIM_ + 255) / 256), 256, 0, stream>>>(X, invx, tokbf);
  k_cvt_pro<<<(int)(((size_t)NPADB * DIM_ + 255) / 256), 256, 0, stream>>>(proto, invp, probf);

  // 2) big WMMA GEMM (TDM double-buffered tiles) -> logits
  k_gemm<<<dim3(16, (M_TOK + 127) / 128), 256, 0, stream>>>(tokbf, probf, scale, outLogits);

  // 3) PCA: column means, power iteration
  k_zerof<<<3, 256, 0, stream>>>(mean, DIM_);
  k_colsum<<<dim3(3, 16), 256, 0, stream>>>(X, mean);
  k_mean_div_initv<<<1, 256, 0, stream>>>(mean, v);
  for (int it = 0; it < 10; ++it) {
    k_pca_prep<<<1, 256, 0, stream>>>(mean, v, vtmp, scal);
    k_matvec_u<<<M_TOK / 8, 256, 0, stream>>>(X, v, scal, u);
    k_matvec_vt<<<dim3(3, 16), 256, 0, stream>>>(X, u, vtmp);
    k_v_finish<<<1, 256, 0, stream>>>(mean, vtmp, scal, v);
  }
  k_pca_prep<<<1, 256, 0, stream>>>(mean, v, vtmp, scal);
  k_matvec_u<<<M_TOK / 8, 256, 0, stream>>>(X, v, scal, u);

  // 4) pseudo labels
  k_minmax_init<<<1, 1, 0, stream>>>(uscal);
  k_minmax<<<(M_TOK + 255) / 256, 256, 0, stream>>>(u, uscal);
  k_thr<<<1, 1, 0, stream>>>(uscal, scal);
  k_pseudo<<<(M_TOK + 255) / 256, 256, 0, stream>>>(u, labels, scal, labint, outPse);

  // 5) sinkhorn on collapsed Q
  k_cls_zero<<<1, 256, 0, stream>>>(Nc, Stot, colsum, flags);
  k_flags<<<1, 32, 0, stream>>>(labels, flags);
  k_qinit<<<(M_TOK + 255) / 256, 256, 0, stream>>>(outLogits, labint, q, Stot, Nc);
  k_qscale<<<(M_TOK + 255) / 256, 256, 0, stream>>>(q, labint, Stot);
  for (int it = 0; it < 3; ++it) {
    k_zerocol<<<1, 256, 0, stream>>>(colsum);
    k_colsumq<<<(M_TOK + 255) / 256, 256, 0, stream>>>(q, labint, colsum);
    k_qrow<<<(M_TOK + 255) / 256, 256, 0, stream>>>(q, labint, colsum, Nc);
  }
  k_qmul<<<(M_TOK + 255) / 256, 256, 0, stream>>>(q, labint, Nc);

  // 6) prototype momentum update
  k_zerof<<<(CC * KK * DIM_ + 255) / 256, 256, 0, stream>>>(Pn, CC * KK * DIM_);
  k_pnew<<<M_TOK / 8, 256, 0, stream>>>(X, invx, labint, q, Pn);
  k_proto_out<<<(CC * KK * DIM_ + 255) / 256, 256, 0, stream>>>(proto, Pn, flags, outPro);

  // 7) pooled heads
  k_imglog<<<dim3(4, BB), 256, 0, stream>>>(outLogits, outImg);
  k_clslog<<<BB, 256, 0, stream>>>(outImg, sa, outCls);
}
